// Q8FQuantLinearFp8_7078106103819
// MI455X (gfx1250) — 8-bit matrix operand order repaired and checked by reading, compiled, not run on hardware
//
#include <hip/hip_runtime.h>
#include <hip/hip_bf16.h>
#include <stdint.h>

typedef __attribute__((ext_vector_type(16))) int   v16i;
typedef __attribute__((ext_vector_type(8)))  float v8f;
typedef int vi4 __attribute__((vector_size(16)));

__device__ __forceinline__ uint32_t f32_to_e4m3_byte(float f) {
    uint32_t u    = __float_as_uint(f);
    uint32_t sign = (u >> 24) & 0x80u;
    uint32_t a    = u & 0x7fffffffu;
    if (a > 0x43E00000u) a = 0x43E00000u;
    uint32_t e = a >> 23;
    uint32_t res;
    if (e > 120u) {
        uint32_t exp8 = e - 120u;
        uint32_t mant = (a >> 20) & 0x7u;
        uint32_t rem  = a & 0xFFFFFu;
        uint32_t v    = (exp8 << 3) | mant;
        if (rem > 0x80000u || (rem == 0x80000u && (v & 1u))) v++;
        if (v > 0x7Eu) v = 0x7Eu;
        res = v;
    } else {
        float k = rintf(__uint_as_float(a) * 512.0f);
        res = (uint32_t)k;
        if (res > 8u) res = 8u;
    }
    return res | sign;
}

__device__ __forceinline__ uint32_t pack4_e4m3(float a, float b, float c, float d) {
    return f32_to_e4m3_byte(a)
         | (f32_to_e4m3_byte(b) << 8)
         | (f32_to_e4m3_byte(c) << 16)
         | (f32_to_e4m3_byte(d) << 24);
}

__global__ __launch_bounds__(256) void quant_x_kernel(const float* __restrict__ x,
                                                      unsigned char* __restrict__ xq,
                                                      float* __restrict__ xs,
                                                      int K) {
    const int row = blockIdx.x;
    const int t   = threadIdx.x;
    const float* xr = x + (size_t)row * K;
    const int K8 = K >> 3;

    float am = 0.0f;
    for (int g = t; g < K8; g += 256) {
        const float4* p = reinterpret_cast<const float4*>(xr) + 2 * g;
        float4 v0 = p[0], v1 = p[1];
        am = fmaxf(am, fmaxf(fmaxf(fabsf(v0.x), fabsf(v0.y)), fmaxf(fabsf(v0.z), fabsf(v0.w))));
        am = fmaxf(am, fmaxf(fmaxf(fabsf(v1.x), fabsf(v1.y)), fmaxf(fabsf(v1.z), fabsf(v1.w))));
    }
    #pragma unroll
    for (int off = 16; off > 0; off >>= 1)
        am = fmaxf(am, __shfl_xor(am, off, 32));

    __shared__ float red[8];
    __shared__ float s_inv;
    if ((t & 31) == 0) red[t >> 5] = am;
    __syncthreads();
    if (t == 0) {
        float m = red[0];
        #pragma unroll
        for (int i = 1; i < 8; ++i) m = fmaxf(m, red[i]);
        float sc = fmaxf(m * (1.0f / 448.0f), 1e-12f);
        xs[row] = sc;
        s_inv   = 1.0f / sc;
    }
    __syncthreads();
    const float inv = s_inv;

    unsigned char* xqr = xq + (size_t)row * K;
    for (int g = t; g < K8; g += 256) {
        const float4* p = reinterpret_cast<const float4*>(xr) + 2 * g;
        float4 v0 = p[0], v1 = p[1];
        uint2 pk;
        pk.x = pack4_e4m3(v0.x * inv, v0.y * inv, v0.z * inv, v0.w * inv);
        pk.y = pack4_e4m3(v1.x * inv, v1.y * inv, v1.z * inv, v1.w * inv);
        reinterpret_cast<uint2*>(xqr)[g] = pk;
    }
}

__global__ __launch_bounds__(256) void quant_w_kernel(const float* __restrict__ w,
                                                      unsigned char* __restrict__ wq,
                                                      size_t total8) {
    size_t i = (size_t)blockIdx.x * 256 + threadIdx.x;
    if (i >= total8) return;
    const float4* p = reinterpret_cast<const float4*>(w) + 2 * i;
    float4 v0 = p[0], v1 = p[1];
    uint2 pk;
    pk.x = pack4_e4m3(v0.x, v0.y, v0.z, v0.w);
    pk.y = pack4_e4m3(v1.x, v1.y, v1.z, v1.w);
    reinterpret_cast<uint2*>(wq)[i] = pk;
}

#if defined(__has_builtin)
#if __has_builtin(__builtin_amdgcn_global_load_async_to_lds_b128)
#define HAVE_ASYNC_LDS 1
#endif
#endif

#ifdef HAVE_ASYNC_LDS
__device__ __forceinline__ void async_cp16(const unsigned char* g, unsigned char* l) {
    __builtin_amdgcn_global_load_async_to_lds_b128(
        (__attribute__((address_space(1))) vi4*)g,
        (__attribute__((address_space(3))) vi4*)l, 0, 0);
}
__device__ __forceinline__ void issue_row128(const unsigned char* g, unsigned char* l) {
    #pragma unroll
    for (int c = 0; c < 8; ++c) async_cp16(g + 16 * c, l + 16 * c);
}
#if __has_builtin(__builtin_amdgcn_s_wait_asynccnt)
#define WAIT_ASYNC(n) __builtin_amdgcn_s_wait_asynccnt(n)
#else
#define WAIT_ASYNC(n) asm volatile("s_wait_asynccnt %0" :: "i"(n))
#endif
#endif

#define LDS_PITCH 144
#define B_OFF     (128 * LDS_PITCH)
#define STAGE_SZ  (2 * 128 * LDS_PITCH)

__global__ __launch_bounds__(256) void gemm_fp8_kernel(
        const unsigned char* __restrict__ Aq,
        const unsigned char* __restrict__ Wq,
        const float* __restrict__ xs,
        const float* __restrict__ wsc,
        const float* __restrict__ bias,
        __hip_bfloat16* __restrict__ out,
        int M, int N, int K) {
#ifdef HAVE_ASYNC_LDS
    __shared__ __align__(16) unsigned char lds[2 * STAGE_SZ];
#else
    __shared__ __align__(16) unsigned char lds[STAGE_SZ];
#endif

    const int lane = threadIdx.x & 31;
    const int wave = threadIdx.x >> 5;
    const int hi   = lane >> 4;
    const int ln   = lane & 15;

    const int wave_m = wave & 3;
    const int wave_n = wave >> 2;
    const int mBlock = blockIdx.y * 128;
    const int nBlock = blockIdx.x * 128;

    const int  crow   = threadIdx.x & 127;
    const bool isB    = threadIdx.x >= 128;
    const unsigned char* grow = isB ? (Wq + (size_t)(nBlock + crow) * K)
                                    : (Aq + (size_t)(mBlock + crow) * K);
    unsigned char* lrow = lds + (isB ? B_OFF : 0) + crow * LDS_PITCH;

    v8f acc[2][4];
    #pragma unroll
    for (int i = 0; i < 2; ++i)
        #pragma unroll
        for (int j = 0; j < 4; ++j)
            acc[i][j] = (v8f){};

    const int NSTEP = K / 128;

    const int aOff = (wave_m * 32 + ln) * LDS_PITCH + 8 * hi;
    const int bOff = B_OFF + (wave_n * 64 + ln) * LDS_PITCH + 8 * hi;

    auto compute_step = [&](const unsigned char* stageBase) {
        v16i a[2];
        #pragma unroll
        for (int i = 0; i < 2; ++i) {
            const unsigned char* p = stageBase + aOff + i * (16 * LDS_PITCH);
            #pragma unroll
            for (int q = 0; q < 8; ++q) {
                int2 t = *reinterpret_cast<const int2*>(p + 16 * q);
                a[i][2 * q] = t.x; a[i][2 * q + 1] = t.y;
            }
        }
        v16i b[4];
        #pragma unroll
        for (int j = 0; j < 4; ++j) {
            const unsigned char* p = stageBase + bOff + j * (16 * LDS_PITCH);
            #pragma unroll
            for (int q = 0; q < 8; ++q) {
                int2 t = *reinterpret_cast<const int2*>(p + 16 * q);
                b[j][2 * q] = t.x; b[j][2 * q + 1] = t.y;
            }
        }
        #pragma unroll
        for (int i = 0; i < 2; ++i)
            #pragma unroll
            for (int j = 0; j < 4; ++j)
                acc[i][j] = __builtin_amdgcn_wmma_f32_16x16x128_fp8_fp8(
                        a[i], b[j], (short)0, acc[i][j], false, false);
    };

#ifdef HAVE_ASYNC_LDS
    issue_row128(grow, lrow);
    const unsigned char* gnext = grow + 128;
    int s = 0;
    for (; s < NSTEP - 1; ++s) {
        issue_row128(gnext, lrow + ((s + 1) & 1) * STAGE_SZ);
        gnext += 128;
        WAIT_ASYNC(8);
        __syncthreads();
        compute_step(lds + (s & 1) * STAGE_SZ);
        __syncthreads();
    }
    WAIT_ASYNC(0);
    __syncthreads();
    compute_step(lds + (s & 1) * STAGE_SZ);
#else
    uint4 stage[8];
    #pragma unroll
    for (int c = 0; c < 8; ++c)
        stage[c] = *reinterpret_cast<const uint4*>(grow + 16 * c);

    for (int s = 0; s < NSTEP; ++s) {
        __syncthreads();
        #pragma unroll
        for (int c = 0; c < 8; ++c)
            *reinterpret_cast<uint4*>(lrow + 16 * c) = stage[c];
        __syncthreads();
        if (s + 1 < NSTEP) {
            const unsigned char* g = grow + (s + 1) * 128;
            #pragma unroll
            for (int c = 0; c < 8; ++c)
                stage[c] = *reinterpret_cast<const uint4*>(g + 16 * c);
        }
        compute_step(lds);
    }
#endif

    const int mBase = mBlock + wave_m * 32;
    const int nBase = nBlock + wave_n * 64;
    #pragma unroll
    for (int i = 0; i < 2; ++i) {
        const int m0 = mBase + 16 * i + 8 * hi;
        float xsv[8];
        #pragma unroll
        for (int r = 0; r < 8; ++r) xsv[r] = xs[m0 + r];
        #pragma unroll
        for (int j = 0; j < 4; ++j) {
            const int   n  = nBase + 16 * j + ln;
            const float wn = wsc[n];
            const float bn = bias[n];
            #pragma unroll
            for (int r = 0; r < 8; ++r) {
                float v = acc[i][j][r] * xsv[r] * wn + bn;
                out[(size_t)(m0 + r) * N + n] = __float2bfloat16(v);
            }
        }
    }
}

extern "C" void kernel_launch(void* const* d_in, const int* in_sizes, int n_in,
                              void* d_out, int out_size, void* d_ws, size_t ws_size,
                              hipStream_t stream) {
    const float* x     = (const float*)d_in[0];
    const float* wq_f  = (const float*)d_in[1];
    const float* wsc   = (const float*)d_in[2];
    const float* bias  = (const float*)d_in[3];

    const int N = in_sizes[2];
    const int K = in_sizes[1] / N;
    const int M = in_sizes[0] / K;

    unsigned char* ws = (unsigned char*)d_ws;
    size_t off_wq = ((size_t)M * K + 255) & ~(size_t)255;
    size_t off_xs = ((off_wq + (size_t)N * K) + 255) & ~(size_t)255;
    unsigned char* xq = ws;
    unsigned char* wq = ws + off_wq;
    float*         xsc = (float*)(ws + off_xs);

    quant_x_kernel<<<M, 256, 0, stream>>>(x, xq, xsc, K);

    size_t total8 = (size_t)N * K / 8;
    quant_w_kernel<<<(unsigned)((total8 + 255) / 256), 256, 0, stream>>>(wq_f, wq, total8);

    dim3 grid((unsigned)(N / 128), (unsigned)(M / 128));
    gemm_fp8_kernel<<<grid, 256, 0, stream>>>(xq, wq, xsc, wsc, bias,
                                              (__hip_bfloat16*)d_out, M, N, K);
}
